// Block_30657476559135
// MI455X (gfx1250) — compile-verified, matrix operand order checked by reading, not run on hardware
//
#include <hip/hip_runtime.h>
#include <hip/hip_bf16.h>
#include <math.h>

// ---------------- types ----------------
typedef __bf16 bf16;
typedef __attribute__((ext_vector_type(16))) __bf16 v16bf;
typedef __attribute__((ext_vector_type(8)))  __bf16 v8bf;
typedef __attribute__((ext_vector_type(8)))  float  v8f;

union UBF8 { v8bf v; bf16 h[8]; unsigned int w[4]; };

// problem constants
#define DIMC    768
#define NHEADS  12
#define HD      64
#define WS14    14
#define NTOK    196           // 14*14
#define NPAD    208           // 13 tiles of 16
#define NWIN    50            // 2 * 5 * 5
#define MROWS   (NWIN * NPAD) // 10400
#define TOKENS  8192          // 2*64*64
#define MLPH    3072

// ---------------- WMMA fragment helpers ----------------
// A-fragment (16x32 bf16) per ISA 7.12.2: lane<16 -> row=lane, K {kb..kb+7, kb+16..kb+23}
//                                         lane>=16 -> row=lane-16, K {kb+8..kb+15, kb+24..kb+31}
__device__ __forceinline__ v16bf load_frag(const bf16* base, int stride, int row0, int kb) {
  int lane = threadIdx.x & 31;
  const bf16* p = base + (row0 + (lane & 15)) * stride + kb + ((lane >> 4) << 3);
  union { v16bf v; v8bf h[2]; } u;
  u.h[0] = *(const v8bf*)(p);
  u.h[1] = *(const v8bf*)(p + 16);
  return u.v;
}

__device__ __forceinline__ v8f wmma_bf16(v16bf a, v16bf b, v8f c) {
  return __builtin_amdgcn_wmma_f32_16x16x32_bf16(false, a, false, b, (short)0, c, false, false);
}

// ---------------- elementwise fp32 -> bf16 ----------------
__global__ void cvt_kernel(const float* __restrict__ s, bf16* __restrict__ d, int n) {
  int i = blockIdx.x * 256 + threadIdx.x;
  if (i < n) d[i] = (bf16)s[i];
}

// ---------------- LayerNorm (per 768-dim token row) -> bf16 ----------------
__global__ __launch_bounds__(256) void ln_kernel(const float* __restrict__ x,
                                                 const float* __restrict__ w,
                                                 const float* __restrict__ b,
                                                 bf16* __restrict__ o) {
  __shared__ float r1[256], r2[256];
  size_t row = blockIdx.x;
  const float* xr = x + row * DIMC;
  float s = 0.f, s2 = 0.f;
  for (int c = threadIdx.x; c < DIMC; c += 256) { float v = xr[c]; s += v; s2 += v * v; }
  r1[threadIdx.x] = s; r2[threadIdx.x] = s2;
  __syncthreads();
  for (int st = 128; st > 0; st >>= 1) {
    if ((int)threadIdx.x < st) { r1[threadIdx.x] += r1[threadIdx.x + st]; r2[threadIdx.x] += r2[threadIdx.x + st]; }
    __syncthreads();
  }
  float mu = r1[0] * (1.f / DIMC);
  float var = r2[0] * (1.f / DIMC) - mu * mu;
  float rstd = rsqrtf(var + 1e-5f);
  for (int c = threadIdx.x; c < DIMC; c += 256) {
    float v = (xr[c] - mu) * rstd * w[c] + b[c];
    o[row * DIMC + c] = (bf16)v;
  }
}

// ---------------- window partition: xn1[8192][768] -> xw[10400][768] ----------------
__global__ void win_kernel(const bf16* __restrict__ xn, bf16* __restrict__ xw) {
  int row = blockIdx.x;                 // 0..10399
  int wdw = row / NPAD, t = row % NPAD;
  int bb = wdw / 25, wr = wdw % 25, wh = wr / 5, ww = wr % 5;
  int gh = wh * WS14 + t / WS14, gw = ww * WS14 + t % WS14;
  bool valid = (t < NTOK) && (gh < 64) && (gw < 64);
  const bf16* src = xn + ((size_t)((bb * 64 + gh) * 64 + gw)) * DIMC;
  for (int c = threadIdx.x; c < DIMC; c += 256)
    xw[(size_t)row * DIMC + c] = valid ? src[c] : (bf16)0.f;
}

// ---------------- generic bf16 WMMA GEMM, 128x128 block tile, double-buffered ----------------
// MODE 0: qkv   -> out bf16,  C = A*B + bias
// MODE 1: proj  -> out f32 x1, window-unpartition scatter + shortcut residual
// MODE 2: fc1   -> out bf16,  exact-erf GELU(A*B + bias)
// MODE 3: fc2   -> out f32,   A*B + bias + resid
#define TS 40   // LDS tile row stride in halves: 80B = 16B multiple, conflict-friendly

__device__ __forceinline__ void stage_tiles(const bf16* __restrict__ A, const bf16* __restrict__ B,
                                            bf16* sA, bf16* sB, int mBase, int nBase, int kb,
                                            int M, int N, int K, int tid) {
  // A tile 128x32: 512 chunks of 8 halves (2 per thread)
#pragma unroll
  for (int it = 0; it < 2; ++it) {
    int c = tid + it * 256;
    int r = c >> 2;               // 0..127
    int k8 = (c & 3) << 3;        // 0,8,16,24
    UBF8 u;
    int gr = mBase + r;
    if (gr < M) {
      u.v = *(const v8bf*)(A + (size_t)gr * K + kb + k8);
      if (kb + 64 < K) __builtin_prefetch(A + (size_t)gr * K + kb + 64, 0, 0);
    } else { u.w[0] = u.w[1] = u.w[2] = u.w[3] = 0u; }
    *(v8bf*)(sA + r * TS + k8) = u.v;
  }
  // B tile 32x128 -> transposed scatter sB[n][k]
#pragma unroll
  for (int it = 0; it < 2; ++it) {
    int c = tid + it * 256;
    int kl = c >> 4;              // 0..31
    int n8 = (c & 15) << 3;
    UBF8 u;
    u.v = *(const v8bf*)(B + (size_t)(kb + kl) * N + nBase + n8);
#pragma unroll
    for (int j = 0; j < 8; ++j) sB[(n8 + j) * TS + kl] = u.h[j];
  }
}

template <int MODE>
__global__ __launch_bounds__(256) void gemm_kernel(const bf16* __restrict__ A,
                                                   const bf16* __restrict__ B,
                                                   const float* __restrict__ bias,
                                                   const float* __restrict__ resid,
                                                   void* __restrict__ out,
                                                   int M, int N, int K) {
  __shared__ bf16 sA[2][128 * TS];
  __shared__ bf16 sB[2][128 * TS];
  const int tid  = threadIdx.x;
  const int lane = tid & 31;
  const int wv   = tid >> 5;
  const int mBase = blockIdx.y * 128;
  const int nBase = blockIdx.x * 128;
  const int wm = wv & 3;          // covers m sub-tiles 2*wm, 2*wm+1
  const int wn = wv >> 2;         // 0/1 -> which 64-wide n half
  v8f acc[2][4];
#pragma unroll
  for (int r = 0; r < 2; ++r)
#pragma unroll
    for (int t = 0; t < 4; ++t)
#pragma unroll
      for (int i = 0; i < 8; ++i) acc[r][t][i] = 0.f;

  stage_tiles(A, B, sA[0], sB[0], mBase, nBase, 0, M, N, K, tid);
  __syncthreads();
  int cur = 0;
  for (int kb = 0; kb < K; kb += 32) {
    if (kb + 32 < K)  // overlap next-tile staging with WMMA on current tile
      stage_tiles(A, B, sA[cur ^ 1], sB[cur ^ 1], mBase, nBase, kb + 32, M, N, K, tid);
    v16bf a0 = load_frag(sA[cur], TS, (wm * 2 + 0) * 16, 0);
    v16bf a1 = load_frag(sA[cur], TS, (wm * 2 + 1) * 16, 0);
#pragma unroll
    for (int t = 0; t < 4; ++t) {
      v16bf bfr = load_frag(sB[cur], TS, wn * 64 + t * 16, 0);
      acc[0][t] = wmma_bf16(a0, bfr, acc[0][t]);
      acc[1][t] = wmma_bf16(a1, bfr, acc[1][t]);
    }
    __syncthreads();
    cur ^= 1;
  }

  // epilogue: C layout -> VGPR i holds M=i (lanes<16) / M=i+8 (lanes>=16), N=lane&15
#pragma unroll
  for (int r = 0; r < 2; ++r) {
    int mo = mBase + (wm * 2 + r) * 16 + ((lane >> 4) << 3);
#pragma unroll
    for (int t = 0; t < 4; ++t) {
      int n = nBase + wn * 64 + t * 16 + (lane & 15);
#pragma unroll
      for (int i = 0; i < 8; ++i) {
        int m = mo + i;
        if (m >= M) continue;
        float v = acc[r][t][i] + bias[n];
        if (MODE == 0) {
          ((bf16*)out)[(size_t)m * N + n] = (bf16)v;
        } else if (MODE == 1) {
          int wdw = m / NPAD, tt = m % NPAD;
          if (tt < NTOK) {
            int bb = wdw / 25, wr = wdw % 25, wh = wr / 5, ww = wr % 5;
            int gh = wh * WS14 + tt / WS14, gw = ww * WS14 + tt % WS14;
            if (gh < 64 && gw < 64) {
              size_t tok = (size_t)(bb * 64 + gh) * 64 + gw;
              ((float*)out)[tok * DIMC + n] = v + resid[tok * DIMC + n];
            }
          }
        } else if (MODE == 2) {
          float g = 0.5f * v * (1.f + erff(v * 0.70710678118654752f));
          ((bf16*)out)[(size_t)m * N + n] = (bf16)g;
        } else {
          ((float*)out)[(size_t)m * N + n] = v + resid[(size_t)m * N + n];
        }
      }
    }
  }
}

// ---------------- attention: one block per (window, head) ----------------
#define QK_S 80    // Q/K LDS row stride (halves); 160B = 16B multiple
#define VP_S 232   // Vt/P LDS row stride (halves); 464B = 16B multiple, conflict-free
#define SM_Q  (NPAD * QK_S)
#define SM_K  (NPAD * QK_S)
#define SM_VT (HD * VP_S)
#define SM_P  (NPAD * VP_S)
#define ATTN_SMEM_BYTES ((SM_Q + SM_K + SM_VT + SM_P) * 2 + 2 * (NPAD * 14) * 4)

__global__ __launch_bounds__(256) void attn_kernel(const bf16* __restrict__ qkv,   // [10400][2304]
                                                   const float* __restrict__ rph,  // [27][64]
                                                   const float* __restrict__ rpw,
                                                   bf16* __restrict__ out) {       // [10400][768]
  extern __shared__ char smem[];
  bf16* sQ  = (bf16*)smem;
  bf16* sK  = sQ + SM_Q;
  bf16* sVt = sK + SM_K;                 // transposed V: [HD][VP_S]
  bf16* sP  = sVt + SM_VT;               // probs bf16: [NPAD][VP_S]
  float* sRH = (float*)(sP + SM_P);      // [NPAD][14]
  float* sRW = sRH + NPAD * 14;

  const int w    = blockIdx.x / NHEADS;
  const int h    = blockIdx.x % NHEADS;
  const int tid  = threadIdx.x;
  const int lane = tid & 31;
  const int wv   = tid >> 5;
  const size_t rowBase = (size_t)w * NPAD;
  const bf16* qg = qkv + rowBase * (3 * DIMC) + h * HD;
  const bf16* kg = qg + DIMC;
  const bf16* vg = qg + 2 * DIMC;

  // ---- phase 0: stage Q, K, Vt; zero pads; rel-pos bias ----
  for (int c = tid; c < NPAD * 8; c += 256) {      // 8 chunks of 8 halves per row
    int r = c >> 3, d8 = (c & 7) << 3;
    *(v8bf*)(sQ + r * QK_S + d8) = *(const v8bf*)(qg + (size_t)r * (3 * DIMC) + d8);
    *(v8bf*)(sK + r * QK_S + d8) = *(const v8bf*)(kg + (size_t)r * (3 * DIMC) + d8);
    UBF8 u; u.v = *(const v8bf*)(vg + (size_t)r * (3 * DIMC) + d8);
#pragma unroll
    for (int j = 0; j < 8; ++j) sVt[(d8 + j) * VP_S + r] = u.h[j];
  }
  for (int c = tid; c < HD * 16; c += 256)         // zero Vt key-pad cols 208..223
    sVt[(c >> 4) * VP_S + NPAD + (c & 15)] = (bf16)0.f;
  for (int c = tid; c < NPAD * 16; c += 256)       // zero P key-pad cols 208..223
    sP[(c >> 4) * VP_S + NPAD + (c & 15)] = (bf16)0.f;
  for (int c = tid; c < (NPAD - NTOK) * 14 * 2; c += 256) { // zero pad rows of bias tables
    int which = c >= (NPAD - NTOK) * 14; int cc = which ? c - (NPAD - NTOK) * 14 : c;
    (which ? sRW : sRH)[(NTOK + cc / 14) * 14 + cc % 14] = 0.f;
  }
  for (int t = tid; t < NTOK * 14 * 2; t += 256) { // decomposed rel-pos: q . R
    int which = t >= NTOK * 14;
    int tt = which ? t - NTOK * 14 : t;
    int row = tt / 14, kk = tt % 14;
    int qi = which ? (row % WS14) : (row / WS14);
    const float* rp = (which ? rpw : rph) + (qi - kk + WS14 - 1) * HD;
    const bf16* qrow = qg + (size_t)row * (3 * DIMC);
    float s = 0.f;
    for (int d = 0; d < HD; ++d) s += (float)qrow[d] * rp[d];
    (which ? sRW : sRH)[row * 14 + kk] = s;
  }
  __syncthreads();

  const float scale = 0.125f;   // 64^-0.5
  const int hi8 = (lane >> 4) << 3;
  const int nl  = lane & 15;

  // ---- scores + in-register softmax, per 16-row tile ----
  for (int tile = wv; tile < 13; tile += 8) {
    v8f acc[13];
#pragma unroll
    for (int ct = 0; ct < 13; ++ct)
#pragma unroll
      for (int i = 0; i < 8; ++i) acc[ct][i] = 0.f;
    v16bf a0 = load_frag(sQ, QK_S, tile * 16, 0);
    v16bf a1 = load_frag(sQ, QK_S, tile * 16, 32);
#pragma unroll
    for (int ct = 0; ct < 13; ++ct) {
      v16bf b0 = load_frag(sK, QK_S, ct * 16, 0);
      v16bf b1 = load_frag(sK, QK_S, ct * 16, 32);
      acc[ct] = wmma_bf16(a0, b0, acc[ct]);
      acc[ct] = wmma_bf16(a1, b1, acc[ct]);
    }
    // scale + rel-pos bias + key mask
#pragma unroll
    for (int ct = 0; ct < 13; ++ct) {
      int n = ct * 16 + nl;
#pragma unroll
      for (int i = 0; i < 8; ++i) {
        int m = tile * 16 + i + hi8;
        float v;
        if (n < NTOK) v = acc[ct][i] * scale + sRH[m * 14 + n / WS14] + sRW[m * 14 + n % WS14];
        else          v = -1e30f;
        acc[ct][i] = v;
      }
    }
    // softmax: each row lives in 16 lanes (one half-wave) at vector index i
#pragma unroll
    for (int i = 0; i < 8; ++i) {
      float rmax = -1e30f;
#pragma unroll
      for (int ct = 0; ct < 13; ++ct) rmax = fmaxf(rmax, acc[ct][i]);
      for (int msk = 1; msk <= 8; msk <<= 1) rmax = fmaxf(rmax, __shfl_xor(rmax, msk, 32));
      float rsum = 0.f;
#pragma unroll
      for (int ct = 0; ct < 13; ++ct) { float e = __expf(acc[ct][i] - rmax); acc[ct][i] = e; rsum += e; }
      for (int msk = 1; msk <= 8; msk <<= 1) rsum += __shfl_xor(rsum, msk, 32);
      float inv = 1.f / rsum;
      int m = tile * 16 + i + hi8;
#pragma unroll
      for (int ct = 0; ct < 13; ++ct) sP[m * VP_S + ct * 16 + nl] = (bf16)(acc[ct][i] * inv);
    }
  }
  __syncthreads();

  // ---- O = P x V via WMMA ----
  for (int tile = wv; tile < 13; tile += 8) {
    v8f o[4];
#pragma unroll
    for (int nt = 0; nt < 4; ++nt)
#pragma unroll
      for (int i = 0; i < 8; ++i) o[nt][i] = 0.f;
#pragma unroll
    for (int ks = 0; ks < 7; ++ks) {               // K = 224 (208 real + zeroed pad)
      v16bf a = load_frag(sP, VP_S, tile * 16, ks * 32);
#pragma unroll
      for (int nt = 0; nt < 4; ++nt) {
        v16bf b = load_frag(sVt, VP_S, nt * 16, ks * 32);
        o[nt] = wmma_bf16(a, b, o[nt]);
      }
    }
#pragma unroll
    for (int nt = 0; nt < 4; ++nt)
#pragma unroll
      for (int i = 0; i < 8; ++i) {
        int m = tile * 16 + i + hi8;
        out[(rowBase + m) * DIMC + h * HD + nt * 16 + nl] = (bf16)o[nt][i];
      }
  }
}

// ---------------- host-side orchestration ----------------
extern "C" void kernel_launch(void* const* d_in, const int* in_sizes, int n_in,
                              void* d_out, int out_size, void* d_ws, size_t ws_size,
                              hipStream_t stream) {
  const float* x      = (const float*)d_in[0];
  const float* n1w    = (const float*)d_in[1];
  const float* n1b    = (const float*)d_in[2];
  const float* qkv_w  = (const float*)d_in[3];
  const float* qkv_b  = (const float*)d_in[4];
  const float* proj_w = (const float*)d_in[5];
  const float* proj_b = (const float*)d_in[6];
  const float* rph    = (const float*)d_in[7];
  const float* rpw    = (const float*)d_in[8];
  const float* n2w    = (const float*)d_in[9];
  const float* n2b    = (const float*)d_in[10];
  const float* fc1_w  = (const float*)d_in[11];
  const float* fc1_b  = (const float*)d_in[12];
  const float* fc2_w  = (const float*)d_in[13];
  const float* fc2_b  = (const float*)d_in[14];

  char* wsp = (char*)d_ws;
  auto carve = [&](size_t bytes) { char* p = wsp; wsp += (bytes + 255) & ~(size_t)255; return p; };
  bf16* wq   = (bf16*)carve((size_t)DIMC * 3 * DIMC * 2);
  bf16* wp   = (bf16*)carve((size_t)DIMC * DIMC * 2);
  bf16* w1   = (bf16*)carve((size_t)DIMC * MLPH * 2);
  bf16* w2   = (bf16*)carve((size_t)MLPH * DIMC * 2);
  bf16* xn1  = (bf16*)carve((size_t)TOKENS * DIMC * 2);
  bf16* xw   = (bf16*)carve((size_t)MROWS * DIMC * 2);
  bf16* qkvb = (bf16*)carve((size_t)MROWS * 3 * DIMC * 2);
  bf16* ao   = (bf16*)carve((size_t)MROWS * DIMC * 2);
  float* x1  = (float*)carve((size_t)TOKENS * DIMC * 4);
  bf16* xn2  = (bf16*)carve((size_t)TOKENS * DIMC * 2);
  bf16* hmid = (bf16*)carve((size_t)TOKENS * MLPH * 2);

  // 1) weights -> bf16
  int nq = DIMC * 3 * DIMC, np = DIMC * DIMC, n1 = DIMC * MLPH, n2 = MLPH * DIMC;
  cvt_kernel<<<(nq + 255) / 256, 256, 0, stream>>>(qkv_w, wq, nq);
  cvt_kernel<<<(np + 255) / 256, 256, 0, stream>>>(proj_w, wp, np);
  cvt_kernel<<<(n1 + 255) / 256, 256, 0, stream>>>(fc1_w, w1, n1);
  cvt_kernel<<<(n2 + 255) / 256, 256, 0, stream>>>(fc2_w, w2, n2);

  // 2) LN1 + window partition
  ln_kernel<<<TOKENS, 256, 0, stream>>>(x, n1w, n1b, xn1);
  win_kernel<<<MROWS, 256, 0, stream>>>(xn1, xw);

  // 3) QKV GEMM: [10400 x 768] x [768 x 2304]
  gemm_kernel<0><<<dim3((3 * DIMC) / 128, (MROWS + 127) / 128), 256, 0, stream>>>(
      xw, wq, qkv_b, nullptr, qkvb, MROWS, 3 * DIMC, DIMC);

  // 4) windowed attention, one block per (window, head)
  attn_kernel<<<NWIN * NHEADS, 256, ATTN_SMEM_BYTES, stream>>>(qkvb, rph, rpw, ao);

  // 5) proj GEMM + unpartition + shortcut residual -> x1 (f32)
  gemm_kernel<1><<<dim3(DIMC / 128, (MROWS + 127) / 128), 256, 0, stream>>>(
      ao, wp, proj_b, x, x1, MROWS, DIMC, DIMC);

  // 6) LN2
  ln_kernel<<<TOKENS, 256, 0, stream>>>(x1, n2w, n2b, xn2);

  // 7) MLP fc1 (+GELU) and fc2 (+residual -> d_out)
  gemm_kernel<2><<<dim3(MLPH / 128, TOKENS / 128), 256, 0, stream>>>(
      xn2, w1, fc1_b, nullptr, hmid, TOKENS, MLPH, DIMC);
  gemm_kernel<3><<<dim3(DIMC / 128, TOKENS / 128), 256, 0, stream>>>(
      hmid, w2, fc2_b, x1, (float*)d_out, TOKENS, DIMC, MLPH);

  (void)in_sizes; (void)n_in; (void)out_size; (void)ws_size;
}
